// Mutation_TAGCN_12232066859620
// MI455X (gfx1250) — matrix operand order repaired and checked by reading, not compiled, not run on hardware
//
#include <hip/hip_runtime.h>
#include <stdint.h>

#define N_NODES 100000
#define N_EDGES 1600000
#define F_IN    128
#define HID     128
#define CLS     40
#define CLS_PAD 48   // pad logits to 3 col-tiles of 16

typedef __bf16 bf16_t;
typedef __attribute__((ext_vector_type(16))) __bf16 v16bf;
typedef __attribute__((ext_vector_type(8)))  float  v8f;

// ---------------------------------------------------------------- utilities

__global__ void fill_zero_kernel(float* __restrict__ p, size_t n) {
    size_t i = (size_t)blockIdx.x * blockDim.x + threadIdx.x;
    size_t stride = (size_t)gridDim.x * blockDim.x;
    for (; i < n; i += stride) p[i] = 0.0f;
}

__global__ void deg_kernel(const int* __restrict__ col, float* __restrict__ deg, int e) {
    int i = blockIdx.x * blockDim.x + threadIdx.x;
    if (i < e)
        __hip_atomic_fetch_add(&deg[col[i]], 1.0f, __ATOMIC_RELAXED, __HIP_MEMORY_SCOPE_AGENT);
}

// in-place: deg -> D^{-1/2} (0 where deg==0)
__global__ void dis_kernel(float* __restrict__ deg, int n) {
    int i = blockIdx.x * blockDim.x + threadIdx.x;
    if (i < n) {
        float d = deg[i];
        deg[i] = (d > 0.0f) ? rsqrtf(fmaxf(d, 1.0f)) : 0.0f;
    }
}

__global__ void norm_kernel(const int* __restrict__ row, const int* __restrict__ col,
                            const float* __restrict__ dis, float* __restrict__ norm, int e) {
    int i = blockIdx.x * blockDim.x + threadIdx.x;
    if (i < e) norm[i] = dis[row[i]] * dis[col[i]];
}

// ------------------------------------------------ weight fragment pre-pack
// Pack W[k][K=128][realCols] (f32, row-major) into per-lane WMMA B fragments
// (bf16). Fragment linear index t = ((wk*colTiles + ct)*4 + kt)*32 + lane;
// each fragment is 16 bf16: element i -> k = kt*32 + (lane/16)*8 + (i&7) + (i/8)*16,
//                                       n = ct*16 + (lane&15)   (0 if n>=realCols)
__global__ void pack_w_kernel(const float* __restrict__ W, bf16_t* __restrict__ Bp,
                              int realCols, int colTiles, int nmats) {
    int t = blockIdx.x * blockDim.x + threadIdx.x;
    int total = nmats * colTiles * 4 * 32;
    if (t >= total) return;
    int lane = t & 31;
    int kt   = (t >> 5) & 3;
    int ct   = (t >> 7) % colTiles;
    int wk   = (t >> 7) / colTiles;
    int n    = ct * 16 + (lane & 15);
    bf16_t* dst = Bp + (size_t)t * 16;
#pragma unroll
    for (int i = 0; i < 16; ++i) {
        int kl = kt * 32 + ((lane >> 4) << 3) + (i & 7) + ((i >> 3) << 4);
        float v = (n < realCols) ? W[((size_t)wk * 128 + kl) * realCols + n] : 0.0f;
        dst[i] = (__bf16)v;
    }
}

// -------------------------------------------------------------------- SpMM
// one wave32 per edge; lane handles 4 consecutive features (float4 gather,
// 4 native f32 scatter-atomics). h is L2-resident (51 MB << 192 MB L2).
__global__ void spmm_kernel(const int* __restrict__ row, const int* __restrict__ col,
                            const float* __restrict__ norm,
                            const float* __restrict__ hin, float* __restrict__ hout, int e) {
    int w = blockIdx.x * (blockDim.x >> 5) + (threadIdx.x >> 5);
    int lane = threadIdx.x & 31;
    if (w >= e) return;
    int r = row[w], c = col[w];
    float nv = norm[w];
    const float4 v = *(const float4*)(hin + (size_t)r * HID + lane * 4);
    float* dst = hout + (size_t)c * HID + lane * 4;
    __hip_atomic_fetch_add(dst + 0, v.x * nv, __ATOMIC_RELAXED, __HIP_MEMORY_SCOPE_AGENT);
    __hip_atomic_fetch_add(dst + 1, v.y * nv, __ATOMIC_RELAXED, __HIP_MEMORY_SCOPE_AGENT);
    __hip_atomic_fetch_add(dst + 2, v.z * nv, __ATOMIC_RELAXED, __HIP_MEMORY_SCOPE_AGENT);
    __hip_atomic_fetch_add(dst + 3, v.w * nv, __ATOMIC_RELAXED, __HIP_MEMORY_SCOPE_AGENT);
}

// ------------------------------------------------------------- WMMA GEMM
// C[M x LDC] += A[M x 128 (f32, cast->bf16 on the fly)] @ W (packed bf16 frags)
// One wave owns a 16 x (CT*16) output strip: the A fragment for each K-step is
// loaded & converted ONCE and reused across all CT column tiles (CT*4 WMMAs
// per wave). LDC is compile-time so the RMW epilogue becomes clauses of
// immediate-offset loads/stores with one wait, not a serial load/add/store chain.
template <int CT, int LDC>
__global__ void __launch_bounds__(128)
gemm_acc_kernel(const float* __restrict__ A, const bf16_t* __restrict__ Bp,
                float* __restrict__ C, int rowTiles) {
    int lane = threadIdx.x & 31;
    int rowTile = blockIdx.x * 4 + (threadIdx.x >> 5);
    if (rowTile >= rowTiles) return;          // wave-uniform exit: EXEC stays all-1s
    int half = lane >> 4;
    int mrow = (rowTile << 4) + (lane & 15);

    v8f zero = {0.f, 0.f, 0.f, 0.f, 0.f, 0.f, 0.f, 0.f};
    v8f acc[CT];
#pragma unroll
    for (int ct = 0; ct < CT; ++ct) acc[ct] = zero;

#pragma unroll
    for (int kt = 0; kt < 4; ++kt) {
        // A fragment (ISA 16-bit A 16x32 layout): elems 0..7 -> K = half*8+0..7,
        //                                         elems 8..15 -> K = 16+half*8+0..7
        const float* ap = A + (size_t)mrow * 128 + kt * 32 + half * 8;
        float t[16];
        float4 f;
        f = *(const float4*)(ap + 0);  t[0] = f.x; t[1] = f.y; t[2]  = f.z; t[3]  = f.w;
        f = *(const float4*)(ap + 4);  t[4] = f.x; t[5] = f.y; t[6]  = f.z; t[7]  = f.w;
        f = *(const float4*)(ap + 16); t[8] = f.x; t[9] = f.y; t[10] = f.z; t[11] = f.w;
        f = *(const float4*)(ap + 20); t[12] = f.x; t[13] = f.y; t[14] = f.z; t[15] = f.w;
        v16bf av;
#pragma unroll
        for (int i = 0; i < 16; ++i) av[i] = (__bf16)t[i];

        // sweep all column tiles with this A fragment (B frags are L0-resident)
#pragma unroll
        for (int ct = 0; ct < CT; ++ct) {
            v16bf bv = *(const v16bf*)(Bp + ((size_t)(ct * 4 + kt) * 32 + lane) * 16);
            acc[ct] = __builtin_amdgcn_wmma_f32_16x16x32_bf16(
                false, av, false, bv, (short)0, acc[ct], false, false);
        }
    }

    // C/D layout: VGPR v -> row = rowTile*16 + half*8 + v, col = ct*16 + (lane&15)
    // Batched RMW: all 8 loads (immediate offsets) -> one wait -> add+store clause.
    float* cbase = C + (size_t)((rowTile << 4) + half * 8) * LDC + (lane & 15);
#pragma unroll
    for (int ct = 0; ct < CT; ++ct) {
        float* crow = cbase + ct * 16;
        float o[8];
#pragma unroll
        for (int v = 0; v < 8; ++v) o[v] = crow[v * LDC];
#pragma unroll
        for (int v = 0; v < 8; ++v) crow[v * LDC] = o[v] + acc[ct][v];
    }
}

// --------------------------------------------------------------- epilogues

__global__ void bias_relu_kernel(float* __restrict__ h, const float* __restrict__ b,
                                 size_t total) {
    size_t i = (size_t)blockIdx.x * blockDim.x + threadIdx.x;
    size_t stride = (size_t)gridDim.x * blockDim.x;
    for (; i < total; i += stride)
        h[i] = fmaxf(h[i] + b[i & (HID - 1)], 0.0f);
}

// one wave per row: add b2, log_softmax over 40 classes (lane + lane+32)
__global__ void logsoftmax_kernel(const float* __restrict__ logits,
                                  const float* __restrict__ b2,
                                  float* __restrict__ out, int n) {
    int w = blockIdx.x * (blockDim.x >> 5) + (threadIdx.x >> 5);
    int lane = threadIdx.x & 31;
    if (w >= n) return;
    const float* lr = logits + (size_t)w * CLS_PAD;
    float v0 = (lane < CLS) ? lr[lane] + b2[lane] : -3.0e38f;
    float v1 = (lane + 32 < CLS) ? lr[lane + 32] + b2[lane + 32] : -3.0e38f;
    float m = fmaxf(v0, v1);
#pragma unroll
    for (int off = 16; off > 0; off >>= 1) m = fmaxf(m, __shfl_xor(m, off, 32));
    float s = __expf(v0 - m) + __expf(v1 - m);
#pragma unroll
    for (int off = 16; off > 0; off >>= 1) s += __shfl_xor(s, off, 32);
    float lse = m + __logf(s);
    if (lane < CLS)      out[(size_t)w * CLS + lane]      = v0 - lse;
    if (lane + 32 < CLS) out[(size_t)w * CLS + lane + 32] = v1 - lse;
}

// ------------------------------------------------------------------ driver

extern "C" void kernel_launch(void* const* d_in, const int* in_sizes, int n_in,
                              void* d_out, int out_size, void* d_ws, size_t ws_size,
                              hipStream_t stream) {
    const float* x  = (const float*)d_in[0];
    const int*   ei = (const int*)d_in[1];      // (2,E) int32
    const float* w1 = (const float*)d_in[2];    // (4,128,128)
    const float* b1 = (const float*)d_in[3];    // (128,)
    const float* w2 = (const float*)d_in[4];    // (4,128,40)
    const float* b2 = (const float*)d_in[5];    // (40,)
    float* out = (float*)d_out;                 // (N,40) log-probs

    const int* row = ei;
    const int* col = ei + N_EDGES;

    // carve workspace (256B aligned)
    char* p = (char*)d_ws;
    auto carve = [&](size_t bytes) -> void* {
        uintptr_t a = (uintptr_t)p;
        a = (a + 255) & ~(uintptr_t)255;
        p = (char*)(a + bytes);
        return (void*)a;
    };
    float*  deg    = (float*)carve(sizeof(float) * N_NODES);
    float*  norm   = (float*)carve(sizeof(float) * N_EDGES);
    float*  hA     = (float*)carve(sizeof(float) * (size_t)N_NODES * HID);
    float*  hB     = (float*)carve(sizeof(float) * (size_t)N_NODES * HID);
    float*  out1   = (float*)carve(sizeof(float) * (size_t)N_NODES * HID);
    float*  logits = (float*)carve(sizeof(float) * (size_t)N_NODES * CLS_PAD);
    bf16_t* w1p    = (bf16_t*)carve(sizeof(bf16_t) * 4 * (HID / 16) * 4 * 32 * 16);
    bf16_t* w2p    = (bf16_t*)carve(sizeof(bf16_t) * 4 * (CLS_PAD / 16) * 4 * 32 * 16);
    (void)ws_size; (void)in_sizes; (void)n_in; (void)out_size;

    const int rowTiles = N_NODES / 16;                       // 6250 exactly
    const size_t wfrag1 = (size_t)(HID / 16) * 4 * 32 * 16;  // frag elems per W1_k
    const size_t wfrag2 = (size_t)(CLS_PAD / 16) * 4 * 32 * 16;

    // ---- gcn_norm
    fill_zero_kernel<<<512, 256, 0, stream>>>(deg, (size_t)N_NODES);
    deg_kernel<<<(N_EDGES + 255) / 256, 256, 0, stream>>>(col, deg, N_EDGES);
    dis_kernel<<<(N_NODES + 255) / 256, 256, 0, stream>>>(deg, N_NODES);
    norm_kernel<<<(N_EDGES + 255) / 256, 256, 0, stream>>>(row, col, deg, norm, N_EDGES);

    // ---- pre-pack weights into WMMA B fragments (bf16)
    pack_w_kernel<<<(4 * (HID / 16) * 4 * 32 + 255) / 256, 256, 0, stream>>>(
        w1, w1p, HID, HID / 16, 4);
    pack_w_kernel<<<(4 * (CLS_PAD / 16) * 4 * 32 + 255) / 256, 256, 0, stream>>>(
        w2, w2p, CLS, CLS_PAD / 16, 4);

    dim3 gg((rowTiles + 3) / 4);

    // ---- layer 1: out1 = sum_k (S^k x) @ W1_k
    fill_zero_kernel<<<4096, 256, 0, stream>>>(out1, (size_t)N_NODES * HID);
    gemm_acc_kernel<8, HID><<<gg, 128, 0, stream>>>(x, w1p, out1, rowTiles);

    const float* hcur = x;
    float* hnext = hA;
    for (int k = 1; k <= 3; ++k) {
        fill_zero_kernel<<<4096, 256, 0, stream>>>(hnext, (size_t)N_NODES * HID);
        spmm_kernel<<<N_EDGES / 8, 256, 0, stream>>>(row, col, norm, hcur, hnext, N_EDGES);
        gemm_acc_kernel<8, HID><<<gg, 128, 0, stream>>>(hnext, w1p + (size_t)k * wfrag1,
                                                        out1, rowTiles);
        hcur = hnext;
        hnext = (hnext == hA) ? hB : hA;
    }
    bias_relu_kernel<<<4096, 256, 0, stream>>>(out1, b1, (size_t)N_NODES * HID);

    // ---- layer 2: logits = sum_k (S^k h) @ W2_k   (LDC = 48, cols 40..47 padded)
    fill_zero_kernel<<<4096, 256, 0, stream>>>(logits, (size_t)N_NODES * CLS_PAD);
    gemm_acc_kernel<3, CLS_PAD><<<gg, 128, 0, stream>>>(out1, w2p, logits, rowTiles);

    hcur = out1;
    hnext = hA;
    for (int k = 1; k <= 3; ++k) {
        fill_zero_kernel<<<4096, 256, 0, stream>>>(hnext, (size_t)N_NODES * HID);
        spmm_kernel<<<N_EDGES / 8, 256, 0, stream>>>(row, col, norm, hcur, hnext, N_EDGES);
        gemm_acc_kernel<3, CLS_PAD><<<gg, 128, 0, stream>>>(hnext, w2p + (size_t)k * wfrag2,
                                                            logits, rowTiles);
        hcur = hnext;
        hnext = (hnext == hA) ? hB : hA;
    }

    // ---- bias + log_softmax -> d_out
    logsoftmax_kernel<<<(N_NODES + 7) / 8, 256, 0, stream>>>(logits, b2, out, N_NODES);
}
